// StandardMultiHeadAttention_8641474200376
// MI455X (gfx1250) — matrix operand order repaired and checked by reading, not compiled, not run on hardware
//
#include <hip/hip_runtime.h>

// ---------------------------------------------------------------------------
// Problem constants (from reference): B=4, S=2048, D=1024, H=16, HD=64
// ---------------------------------------------------------------------------
#define BB   4
#define SS   2048
#define DD   1024
#define HH   16
#define HDIM 64
#define MM   (BB * SS)      // 8192 token rows

typedef __attribute__((ext_vector_type(16))) __bf16 v16bf;
typedef __attribute__((ext_vector_type(8)))  float  v8f;

// f32 -> bf16, round-to-nearest-even
__device__ __forceinline__ unsigned short f2bf_bits(float f) {
  unsigned u = __builtin_bit_cast(unsigned, f);
  u += 0x7FFFu + ((u >> 16) & 1u);
  return (unsigned short)(u >> 16);
}
__device__ __forceinline__ __bf16 f2bf(float f) {
  unsigned short h = f2bf_bits(f);
  return __builtin_bit_cast(__bf16, h);
}

// A-matrix fragment (16x32 bf16), per ISA table: lanes 0-15 hold the row's
// K=[0..7] and K=[16..23]; lanes 16-31 hold K=[8..15] and K=[24..31].
// p must point at (row_base + cA) where cA = (lane>=16) ? 8 : 0.
__device__ __forceinline__ v16bf load_fragA(const __bf16* p) {
  union { uint4 q[2]; v16bf v; } u;
  u.q[0] = *reinterpret_cast<const uint4*>(p);
  u.q[1] = *reinterpret_cast<const uint4*>(p + 16);
  return u.v;
}

// B-matrix fragment (32x16 bf16), same k order as A: lane n (0-15) holds
// column n, K=[0..7] and K=[16..23]; lanes 16-31 hold K=[8..15] and K=[24..31].
// p must point at (col_base + cB) where cB = (lane>=16) ? 8 : 0.
__device__ __forceinline__ v16bf load_fragB(const __bf16* p) {
  union { uint4 q[2]; v16bf v; } u;
  u.q[0] = *reinterpret_cast<const uint4*>(p);
  u.q[1] = *reinterpret_cast<const uint4*>(p + 16);
  return u.v;
}

__device__ __forceinline__ v8f wmma_bf16(v16bf a, v16bf b, v8f c) {
  return __builtin_amdgcn_wmma_f32_16x16x32_bf16(
      /*neg_a=*/false, a, /*neg_b=*/false, b,
      /*c_mod=*/(short)0, c, /*reuse_a=*/false, /*reuse_b=*/false);
}

// 16-lane (half-wave) reductions; xor masks 1,2,4,8 stay inside each half.
__device__ __forceinline__ float rmax16(float v) {
#pragma unroll
  for (int m = 1; m < 16; m <<= 1) v = fmaxf(v, __shfl_xor(v, m, 32));
  return v;
}
__device__ __forceinline__ float rsum16(float v) {
#pragma unroll
  for (int m = 1; m < 16; m <<= 1) v += __shfl_xor(v, m, 32);
  return v;
}

// ---------------------------------------------------------------------------
// f32 -> bf16 conversion (4 elems/thread)
// ---------------------------------------------------------------------------
__global__ void cvt_bf16_kernel(const float* __restrict__ in,
                                __bf16* __restrict__ out, int n4) {
  int i = blockIdx.x * blockDim.x + threadIdx.x;
  if (i < n4) {
    float4 f = reinterpret_cast<const float4*>(in)[i];
    ushort4 o;
    o.x = f2bf_bits(f.x); o.y = f2bf_bits(f.y);
    o.z = f2bf_bits(f.z); o.w = f2bf_bits(f.w);
    reinterpret_cast<ushort4*>(out)[i] = o;
  }
}

// ---------------------------------------------------------------------------
// WMMA GEMM: C[M,N] = A[M,K] @ W[N,K]^T, then (C + bias[N]) * scale.
// Block = 256 threads = 8 waves; wave tile 32(M) x 64(N); block tile 128x128.
// mode 0: store bf16 to [B,H,S,HD]          (Q, K)
// mode 1: store bf16 to [B,H,HD,S]          (V transposed)
// mode 2: store f32  to [M,N] row-major     (final output)
// ---------------------------------------------------------------------------
__global__ __launch_bounds__(256)
void wmma_gemm_bf16(const __bf16* __restrict__ A, const __bf16* __restrict__ W,
                    const float* __restrict__ bias, void* __restrict__ out,
                    int K, float scale, int mode) {
  const int lane = threadIdx.x & 31;
  const int wid  = threadIdx.x >> 5;
  const int wm   = wid & 3;          // 4 waves along M
  const int wn   = wid >> 2;         // 2 waves along N
  const int mbase = blockIdx.x * 128 + wm * 32;
  const int nbase = blockIdx.y * 128 + wn * 64;
  const int lm = lane & 15;
  const int hi = lane >> 4;
  const int cA = hi << 3;            // A k-offset: 0 or 8
  const int cB = hi << 3;            // B k-offset: 0 or 8

  v8f acc[2][4];
#pragma unroll
  for (int i = 0; i < 2; ++i)
#pragma unroll
    for (int j = 0; j < 4; ++j) acc[i][j] = (v8f)0.0f;

  const __bf16* a0p = A + (size_t)(mbase + lm) * K + cA;
  const __bf16* a1p = a0p + (size_t)16 * K;
  const __bf16* wp[4];
#pragma unroll
  for (int j = 0; j < 4; ++j)
    wp[j] = W + (size_t)(nbase + j * 16 + lm) * K + cB;

  for (int k0 = 0; k0 < K; k0 += 32) {
    v16bf a0 = load_fragA(a0p + k0);
    v16bf a1 = load_fragA(a1p + k0);
#pragma unroll
    for (int j = 0; j < 4; ++j) {
      v16bf b = load_fragB(wp[j] + k0);
      acc[0][j] = wmma_bf16(a0, b, acc[0][j]);
      acc[1][j] = wmma_bf16(a1, b, acc[1][j]);
    }
  }

  // Epilogue.  D layout: col = lm; row = (i*16) + (lane>=16 ? 8:0) + vgpr r.
  const int rbase = hi << 3;
#pragma unroll
  for (int j = 0; j < 4; ++j) {
    const int col = nbase + j * 16 + lm;
    const float bv = bias[col];
    const int h  = col >> 6;       // head
    const int hd = col & 63;       // dim within head
#pragma unroll
    for (int i = 0; i < 2; ++i) {
#pragma unroll
      for (int r = 0; r < 8; ++r) {
        const int row = mbase + i * 16 + rbase + r;   // token index
        const float val = (acc[i][j][r] + bv) * scale;
        if (mode == 2) {
          reinterpret_cast<float*>(out)[(size_t)row * DD + col] = val;
        } else {
          const int b = row >> 11;          // S = 2048
          const int s = row & (SS - 1);
          const int bh = b * HH + h;
          size_t dst;
          if (mode == 0) dst = ((size_t)bh * SS + s) * HDIM + hd;   // [B,H,S,HD]
          else           dst = ((size_t)bh * HDIM + hd) * SS + s;   // [B,H,HD,S]
          reinterpret_cast<__bf16*>(out)[dst] = f2bf(val);
        }
      }
    }
  }
}

// ---------------------------------------------------------------------------
// Causal flash attention: one wave per 16-query tile of one (b,h).
// Q,K bf16 [B,H,S,HD]; Vt bf16 [B,H,HD,S]; O bf16 [B*S, D].
// Q is pre-scaled by 1/sqrt(HD).
// ---------------------------------------------------------------------------
__global__ __launch_bounds__(128)
void flash_attn_kernel(const __bf16* __restrict__ Q, const __bf16* __restrict__ Kt,
                       const __bf16* __restrict__ Vt, __bf16* __restrict__ O) {
  __shared__ __attribute__((aligned(16))) __bf16 lds[4][16 * 32];

  const int lane = threadIdx.x & 31;
  const int wid  = threadIdx.x >> 5;
  const int qblk = blockIdx.x * 4 + wid;          // global 16-row query tile
  const int bh = qblk / (SS / 16);
  const int qb = qblk % (SS / 16);
  const int q0 = qb * 16;
  const int lm = lane & 15;
  const int hi = lane >> 4;
  const int cA = hi << 3;
  const int cB = hi << 3;
  const int rbase = hi << 3;
  __bf16* stage = lds[wid];

  // Q A-fragments for both 32-wide hd chunks (K dim of QK^T is HD=64).
  const __bf16* qp = Q + ((size_t)bh * SS + q0 + lm) * HDIM + cA;
  const v16bf qf0 = load_fragA(qp);
  const v16bf qf1 = load_fragA(qp + 32);

  float mrow[8], lrow[8];
  v8f oacc[4];
#pragma unroll
  for (int r = 0; r < 8; ++r) { mrow[r] = -1e30f; lrow[r] = 0.0f; }
#pragma unroll
  for (int g = 0; g < 4; ++g) oacc[g] = (v8f)0.0f;

  const int nblk   = qb + 1;             // 16-key blocks needed (causal)
  const int nchunk = (nblk + 1) >> 1;    // processed as 32-key chunks

  for (int t = 0; t < nchunk; ++t) {
    const int kk = t * 32;

    // ---- scores: S = Q @ K^T (two 16-key sub-blocks) ----
    // B operand lane = key column; hd runs [cB..cB+7] and [cB+16..cB+23] per chunk.
    const __bf16* kp0 = Kt + ((size_t)bh * SS + kk + lm) * HDIM + cB;
    v8f s0 = (v8f)0.0f;
    s0 = wmma_bf16(qf0, load_fragB(kp0), s0);
    s0 = wmma_bf16(qf1, load_fragB(kp0 + 32), s0);

    const bool has1 = (kk + 16) < nblk * 16;   // wave-uniform
    v8f s1;
    if (has1) {
      const __bf16* kp1 = Kt + ((size_t)bh * SS + kk + 16 + lm) * HDIM + cB;
      s1 = (v8f)0.0f;
      s1 = wmma_bf16(qf0, load_fragB(kp1), s1);
      s1 = wmma_bf16(qf1, load_fragB(kp1 + 32), s1);
    } else {
      s1 = (v8f)(-1e30f);
    }

    // ---- causal mask (key > query -> -inf) ----
#pragma unroll
    for (int r = 0; r < 8; ++r) {
      const int qrow = q0 + rbase + r;
      if (kk + lm > qrow) s0[r] = -1e30f;
      if (has1 && (kk + 16 + lm > qrow)) s1[r] = -1e30f;
    }

    // ---- online softmax update ----
    float sc[8];
#pragma unroll
    for (int r = 0; r < 8; ++r) {
      float v = rmax16(fmaxf(s0[r], s1[r]));
      float m2 = fmaxf(mrow[r], v);
      sc[r] = __expf(mrow[r] - m2);
      mrow[r] = m2;
    }
#pragma unroll
    for (int g = 0; g < 4; ++g)
#pragma unroll
      for (int r = 0; r < 8; ++r) oacc[g][r] *= sc[r];

    // ---- P = exp(S - m), stash to LDS in bf16, accumulate row sums ----
#pragma unroll
    for (int r = 0; r < 8; ++r) {
      const float p0 = __expf(s0[r] - mrow[r]);
      const float p1 = __expf(s1[r] - mrow[r]);
      lrow[r] = lrow[r] * sc[r] + rsum16(p0 + p1);
      const int m = rbase + r;
      stage[m * 32 + lm]      = f2bf(p0);
      stage[m * 32 + 16 + lm] = f2bf(p1);
    }
    asm volatile("" ::: "memory");   // keep LDS store->load order; in-order DS per wave

    // ---- O += P(16x32) @ V(32x64): A-frag from LDS, B-frags from Vt ----
    const v16bf pf = load_fragA(stage + lm * 32 + cA);
#pragma unroll
    for (int g = 0; g < 4; ++g) {
      const __bf16* vp = Vt + ((size_t)bh * HDIM + g * 16 + lm) * SS + kk + cB;
      oacc[g] = wmma_bf16(pf, load_fragB(vp), oacc[g]);
    }
    asm volatile("" ::: "memory");
  }

  // ---- finalize: O / l, store bf16 flat [B*S, D] ----
  const int b = bh >> 4, h = bh & 15;
#pragma unroll
  for (int g = 0; g < 4; ++g) {
#pragma unroll
    for (int r = 0; r < 8; ++r) {
      const int srow = q0 + rbase + r;
      const int col  = h * HDIM + g * 16 + lm;
      O[((size_t)(b * SS + srow)) * DD + col] = f2bf(oacc[g][r] / lrow[r]);
    }
  }
}

// ---------------------------------------------------------------------------
// Launch
// ---------------------------------------------------------------------------
extern "C" void kernel_launch(void* const* d_in, const int* in_sizes, int n_in,
                              void* d_out, int out_size, void* d_ws, size_t ws_size,
                              hipStream_t stream) {
  (void)in_sizes; (void)n_in; (void)out_size; (void)ws_size;
  const float* x  = (const float*)d_in[0];
  // d_in[1] = mask (causal, known analytically) -- unused
  const float* Wq = (const float*)d_in[2];
  const float* bq = (const float*)d_in[3];
  const float* Wk = (const float*)d_in[4];
  const float* bk = (const float*)d_in[5];
  const float* Wv = (const float*)d_in[6];
  const float* bv = (const float*)d_in[7];
  const float* Wo = (const float*)d_in[8];
  const float* bo = (const float*)d_in[9];
  float* out = (float*)d_out;

  char* ws = (char*)d_ws;
  const size_t SZ_X = (size_t)MM * DD * 2;       // 16 MB bf16
  const size_t SZ_W = (size_t)DD * DD * 2;       //  2 MB bf16
  __bf16* xb  = (__bf16*)(ws);
  __bf16* Wqb = (__bf16*)(ws + SZ_X);
  __bf16* Wkb = (__bf16*)(ws + SZ_X + SZ_W);
  __bf16* Wvb = (__bf16*)(ws + SZ_X + 2 * SZ_W);
  __bf16* Wob = (__bf16*)(ws + SZ_X + 3 * SZ_W);
  __bf16* Qb  = (__bf16*)(ws + SZ_X + 4 * SZ_W);
  __bf16* Kb  = (__bf16*)(ws + 2 * SZ_X + 4 * SZ_W);
  __bf16* Vtb = (__bf16*)(ws + 3 * SZ_X + 4 * SZ_W);
  __bf16* Ob  = (__bf16*)(ws + 4 * SZ_X + 4 * SZ_W);

  // 1) fp32 -> bf16 conversions
  {
    int n4 = MM * DD / 4;
    cvt_bf16_kernel<<<(n4 + 255) / 256, 256, 0, stream>>>(x, xb, n4);
    int w4 = DD * DD / 4;
    cvt_bf16_kernel<<<(w4 + 255) / 256, 256, 0, stream>>>(Wq, Wqb, w4);
    cvt_bf16_kernel<<<(w4 + 255) / 256, 256, 0, stream>>>(Wk, Wkb, w4);
    cvt_bf16_kernel<<<(w4 + 255) / 256, 256, 0, stream>>>(Wv, Wvb, w4);
    cvt_bf16_kernel<<<(w4 + 255) / 256, 256, 0, stream>>>(Wo, Wob, w4);
  }

  // 2) Q/K/V projections (Q pre-scaled by 1/sqrt(HD)=0.125; V stored transposed)
  dim3 ggrid(MM / 128, DD / 128);
  wmma_gemm_bf16<<<ggrid, 256, 0, stream>>>(xb, Wqb, bq, Qb,  DD, 0.125f, 0);
  wmma_gemm_bf16<<<ggrid, 256, 0, stream>>>(xb, Wkb, bk, Kb,  DD, 1.0f,   0);
  wmma_gemm_bf16<<<ggrid, 256, 0, stream>>>(xb, Wvb, bv, Vtb, DD, 1.0f,   1);

  // 3) causal flash attention -> Ob (bf16 [B*S, D])
  flash_attn_kernel<<<(BB * HH * (SS / 16)) / 4, 128, 0, stream>>>(Qb, Kb, Vtb, Ob);

  // 4) output projection -> d_out (fp32)
  wmma_gemm_bf16<<<ggrid, 256, 0, stream>>>(Ob, Wob, bo, out, DD, 1.0f, 2);
}
